// AttnBlock_5171140624620
// MI455X (gfx1250) — matrix operand order repaired and checked by reading, not compiled, not run on hardware
//
#include <hip/hip_runtime.h>
#include <hip/hip_bf16.h>
#include <math.h>

// ---------------------------------------------------------------------------
// AttnBlock for MI455X (gfx1250): GroupNorm + QKV proj + flash attention +
// out-proj + residual. GEMMs via v_wmma_f32_16x16x32_bf16 (wave32); V tiles
// staged into LDS by the Tensor Data Mover (tensor_load_to_lds / TENSORcnt).
// ---------------------------------------------------------------------------

typedef __attribute__((ext_vector_type(16))) __bf16 v16bf;
typedef __attribute__((ext_vector_type(8)))  __bf16 v8bf;
typedef __attribute__((ext_vector_type(8)))  float  v8f;
typedef __attribute__((ext_vector_type(4)))  unsigned int u32x4;
typedef __attribute__((ext_vector_type(8)))  int i32x8;
typedef __attribute__((ext_vector_type(4)))  int i32x4;

#define C_DIM 512
#define N_SP  4096          // 64*64 spatial positions
#define BATCH 4
#define NGRP  32
#define CPG   16            // channels per group
#define NTOT  (BATCH * N_SP)
#define KBLK  32            // attention key block

// ---------------- WMMA fragment loaders (wave32, contiguous k order shared by A and B) ------

// A matrix 16x32 bf16, source row-major with leading dim lda.
// lane<16: row M=lane, K = 0..15 ; lane>=16: row M=lane-16, K = 16..31
__device__ __forceinline__ v16bf load_a16x32(const __bf16* base, int lda) {
  const int lane = threadIdx.x & 31;
  const int m    = lane & 15;
  const int k0   = (lane < 16) ? 0 : 16;
  const __bf16* p = base + m * lda + k0;
  v8bf lo = *(const v8bf*)(p);
  v8bf hi = *(const v8bf*)(p + 8);
  v16bf a;
#pragma unroll
  for (int i = 0; i < 8; ++i) { a[i] = lo[i]; a[8 + i] = hi[i]; }
  return a;
}

// B matrix 32x16 bf16 where element (k,n) = src[n*ldb + k]  (i.e. B^T stored
// row-major: weight [O][C] rows, or K-matrix [keys][C] rows -> contiguous K).
// lane holds column n = lane&15; lanes 0-15: K=0..15, lanes 16-31: K=16..31.
__device__ __forceinline__ v16bf load_b_rows(const __bf16* base, int ldb) {
  const int lane = threadIdx.x & 31;
  const int n    = lane & 15;
  const int k0   = (lane < 16) ? 0 : 16;
  const __bf16* p = base + n * ldb + k0;
  v8bf lo = *(const v8bf*)(p);
  v8bf hi = *(const v8bf*)(p + 8);
  v16bf b;
#pragma unroll
  for (int i = 0; i < 8; ++i) { b[i] = lo[i]; b[8 + i] = hi[i]; }
  return b;
}

// B matrix 32x16 bf16 where element (k,n) = src[k*ld + n]  (V tiles in LDS).
__device__ __forceinline__ v16bf load_b_strided(const __bf16* base, int ld) {
  const int lane = threadIdx.x & 31;
  const int n    = lane & 15;
  const int k0   = (lane < 16) ? 0 : 16;
  v16bf b;
#pragma unroll
  for (int i = 0; i < 16; ++i) b[i] = base[(k0 + i) * ld + n];
  return b;
}

// ---------------- Tensor Data Mover: 2-D tile (KBLK rows x 512 bf16) --------
// D# packing per CDNA5 ISA 8.3/8.4: count=1, data_size=2B, type=2 ("image"),
// workgroup_mask=0 (not in a cluster). lds_byte_off is the LDS byte address
// of the destination tile (V tile is the first __shared__ object -> offset 0).
__device__ __forceinline__ void tdm_load_tile(const __bf16* gsrc,
                                              unsigned lds_byte_off) {
  unsigned long long ga = (unsigned long long)(uintptr_t)gsrc;
  u32x4 g0 = { 1u,                                   // count=1, user mode
               lds_byte_off,                         // lds_addr
               (unsigned)(ga & 0xFFFFFFFFull),       // global_addr[31:0]
               (unsigned)((ga >> 32) & 0x01FFFFFFull) | 0x80000000u }; // +type=2
  i32x8 g1 = { (int)0x00010000u,        // data_size=1 (2 bytes)
               (int)(512u << 16),       // tensor_dim0 = 512 (elems, contiguous)
               (int)((unsigned)KBLK << 16), // tensor_dim1 = KBLK rows
               (int)(512u << 16),       // tile_dim0 = 512
               (int)KBLK,               // tile_dim1 = KBLK
               512,                     // tensor_dim0_stride = 512 elems
               0, 0 };
  i32x4 z4 = { 0, 0, 0, 0 };
#if __has_include(<hip/amd_detail/amd_gfx1250_TDM.h>)
  i32x8 z8 = { 0, 0, 0, 0, 0, 0, 0, 0 };
  __builtin_amdgcn_tensor_load_to_lds(g0, g1, z4, z4, z8, 0);
#else
  __builtin_amdgcn_tensor_load_to_lds(g0, g1, z4, z4, 0);
#endif
}

// ---------------- GroupNorm statistics --------------------------------------
__global__ __launch_bounds__(256) void gn_stats_kernel(
    const float* __restrict__ x, float* __restrict__ stats) {
  __shared__ float s1[256], s2[256];
  const int bg = blockIdx.x;
  const float4* p = (const float4*)(x + (size_t)bg * CPG * N_SP);
  float a = 0.f, q = 0.f;
  for (int i = threadIdx.x; i < (CPG * N_SP) / 4; i += 256) {
    float4 v = p[i];
    a += v.x + v.y + v.z + v.w;
    q += v.x * v.x + v.y * v.y + v.z * v.z + v.w * v.w;
  }
  const int t = threadIdx.x;
  s1[t] = a; s2[t] = q;
  __syncthreads();
  for (int off = 128; off > 0; off >>= 1) {
    if (t < off) { s1[t] += s1[t + off]; s2[t] += s2[t + off]; }
    __syncthreads();
  }
  if (t == 0) {
    const float inv_n = 1.0f / (float)(CPG * N_SP);
    float mean = s1[0] * inv_n;
    float var  = s2[0] * inv_n - mean * mean;
    stats[bg * 2]     = mean;
    stats[bg * 2 + 1] = rsqrtf(var + 1e-6f);
  }
}

// ---------------- GroupNorm apply + transpose to bf16 [B][N][C] -------------
__global__ __launch_bounds__(256) void gn_apply_kernel(
    const float* __restrict__ x, const float* __restrict__ stats,
    const float* __restrict__ gamma, const float* __restrict__ beta,
    __bf16* __restrict__ h) {
  __shared__ float tile[32][33];
  const int n0 = blockIdx.x * 32;
  const int c0 = blockIdx.y * 32;
  const int b  = blockIdx.z;
  const int tr = threadIdx.x >> 5;   // 0..7
  const int tc = threadIdx.x & 31;
#pragma unroll
  for (int it = 0; it < 4; ++it) {
    int cl = it * 8 + tr;
    int c  = c0 + cl;
    int g  = c >> 4;
    float mean = stats[(b * NGRP + g) * 2];
    float rstd = stats[(b * NGRP + g) * 2 + 1];
    float val  = x[((size_t)(b * C_DIM + c)) * N_SP + n0 + tc];
    tile[cl][tc] = (val - mean) * rstd * gamma[c] + beta[c];
  }
  __syncthreads();
#pragma unroll
  for (int it = 0; it < 4; ++it) {
    int nl = it * 8 + tr;
    h[((size_t)(b * N_SP) + n0 + nl) * C_DIM + c0 + tc] = (__bf16)tile[tc][nl];
  }
}

// ---------------- fp32 -> bf16 weight convert -------------------------------
__global__ __launch_bounds__(256) void cvt_bf16_kernel(
    const float* __restrict__ w, __bf16* __restrict__ o, int n) {
  int i = blockIdx.x * 256 + threadIdx.x;
  if (i < n) o[i] = (__bf16)w[i];
}

// ---------------- QKV projection GEMM: out[n][o] = sum_c A[n][c]*W[o][c]+b --
__global__ __launch_bounds__(256) void gemm_qkv_kernel(
    const __bf16* __restrict__ A, const __bf16* __restrict__ W,
    const float* __restrict__ bias, __bf16* __restrict__ out) {
  const int wave = threadIdx.x >> 5;
  const int lane = threadIdx.x & 31;
  const int row0 = (blockIdx.x * 8 + wave) * 16;
  const int col0 = blockIdx.y * 16;
  const __bf16* a_ptr = A + (size_t)row0 * C_DIM;
  const __bf16* w_ptr = W + (size_t)col0 * C_DIM;
  v8f acc = {};
#pragma unroll 4
  for (int kk = 0; kk < C_DIM; kk += 32) {
    __builtin_prefetch(a_ptr + kk + 128, 0, 3);   // global_prefetch_b8
    v16bf a = load_a16x32(a_ptr + kk, C_DIM);
    v16bf w = load_b_rows(w_ptr + kk, C_DIM);
    acc = __builtin_amdgcn_wmma_f32_16x16x32_bf16(false, a, false, w,
                                                  (short)0, acc, false, false);
  }
  const int hi = lane >> 4, nc = lane & 15;
  const float bv = bias[col0 + nc];
#pragma unroll
  for (int r = 0; r < 8; ++r)
    out[(size_t)(row0 + r + hi * 8) * C_DIM + col0 + nc] = (__bf16)(acc[r] + bv);
}

// ---------------- Flash attention ------------------------------------------
// One workgroup = 16 query rows; 8 waves each own a 64-wide slice of C=512.
// Per 32-key block:
//   wave0 issues a TDM load of the V tile (32 x 512 bf16) into LDS, then all
//   waves compute partial S via WMMA (ds_add_f32 reduce in LDS), 16 threads
//   run the online softmax, s_wait_tensorcnt gates the V tile, and each wave
//   accumulates P@V into its 16x64 f32 accumulator from LDS fragments.
__global__ __launch_bounds__(256) void attn_kernel(
    const __bf16* __restrict__ q, const __bf16* __restrict__ k,
    const __bf16* __restrict__ v, __bf16* __restrict__ o, float scale) {
  // V tile FIRST so its LDS byte offset is 0 (used by the TDM descriptor).
  __shared__ __align__(16) __bf16 Vt[KBLK * C_DIM];          // 32 KB
  __shared__ __align__(16) float  Sb[16 * KBLK];             // 2 KB
  __shared__ __align__(16) __bf16 Pb[16 * KBLK];             // 1 KB
  __shared__ float m_i[16], l_i[16], alpha_s[16];

  const int tid  = threadIdx.x;
  const int wave = tid >> 5;
  const int lane = tid & 31;
  const int b    = blockIdx.y;
  const int n0   = blockIdx.x * 16;

  const __bf16* qb = q + ((size_t)b * N_SP + n0) * C_DIM;
  const __bf16* kb = k + (size_t)b * N_SP * C_DIM;
  const __bf16* vb = v + (size_t)b * N_SP * C_DIM;

  // preload this wave's Q slice (16 rows x 64 cols) as two A fragments
  v16bf qa0 = load_a16x32(qb + wave * 64,      C_DIM);
  v16bf qa1 = load_a16x32(qb + wave * 64 + 32, C_DIM);

  v8f Oacc[4] = {{}, {}, {}, {}};

  if (tid < 16) { m_i[tid] = -INFINITY; l_i[tid] = 0.f; }

  const int hi = lane >> 4, nc = lane & 15;

  for (int kb0 = 0; kb0 < N_SP; kb0 += KBLK) {
    __syncthreads();                    // V tile WAR: prev PV reads done
    // async DMA of V tile for this key block (overlaps with S compute)
    if (wave == 0) tdm_load_tile(vb + (size_t)kb0 * C_DIM, 0u);

    // zero the shared score tile
    for (int i = tid; i < 16 * KBLK; i += 256) Sb[i] = 0.f;
    __syncthreads();

    // partial S: this wave's 64-dim slice vs 2 key sub-tiles of 16
#pragma unroll
    for (int sub = 0; sub < 2; ++sub) {
      const __bf16* kbase = kb + (size_t)(kb0 + sub * 16) * C_DIM + wave * 64;
      v8f acc = {};
      acc = __builtin_amdgcn_wmma_f32_16x16x32_bf16(
          false, qa0, false, load_b_rows(kbase, C_DIM), (short)0, acc, false, false);
      acc = __builtin_amdgcn_wmma_f32_16x16x32_bf16(
          false, qa1, false, load_b_rows(kbase + 32, C_DIM), (short)0, acc, false, false);
#pragma unroll
      for (int r = 0; r < 8; ++r)
        atomicAdd(&Sb[(r + hi * 8) * KBLK + sub * 16 + nc], acc[r]);  // ds_add_f32
    }
    __syncthreads();

    // online softmax: thread t owns query row t
    if (tid < 16) {
      float mo = m_i[tid];
      float rm = mo;
      for (int j = 0; j < KBLK; ++j) rm = fmaxf(rm, Sb[tid * KBLK + j] * scale);
      float al = __expf(mo - rm);
      float ls = 0.f;
      for (int j = 0; j < KBLK; ++j) {
        float p = __expf(Sb[tid * KBLK + j] * scale - rm);
        Pb[tid * KBLK + j] = (__bf16)p;
        ls += p;
      }
      m_i[tid]     = rm;
      l_i[tid]     = l_i[tid] * al + ls;
      alpha_s[tid] = al;
    }
    if (wave == 0) __builtin_amdgcn_s_wait_tensorcnt(0);  // V tile landed
    __syncthreads();

    // rescale running O, then O += P @ V_block (this wave's 64-wide C slice)
#pragma unroll
    for (int r = 0; r < 8; ++r) {
      float av = alpha_s[r + hi * 8];
      Oacc[0][r] *= av; Oacc[1][r] *= av; Oacc[2][r] *= av; Oacc[3][r] *= av;
    }
    v16bf p0 = load_a16x32(&Pb[0], KBLK);
#pragma unroll
    for (int t = 0; t < 4; ++t) {
      const __bf16* vbase = &Vt[wave * 64 + t * 16];
      Oacc[t] = __builtin_amdgcn_wmma_f32_16x16x32_bf16(
          false, p0, false, load_b_strided(vbase, C_DIM),
          (short)0, Oacc[t], false, false);
    }
  }

  // final normalization and store bf16 [B][N][C]
#pragma unroll
  for (int t = 0; t < 4; ++t) {
#pragma unroll
    for (int r = 0; r < 8; ++r) {
      float li  = l_i[r + hi * 8];
      float val = Oacc[t][r] / li;
      o[((size_t)b * N_SP + n0 + r + hi * 8) * C_DIM + wave * 64 + t * 16 + nc] =
          (__bf16)val;
    }
  }
}

// ---------------- Output projection + bias + residual ----------------------
__global__ __launch_bounds__(256) void oproj_kernel(
    const __bf16* __restrict__ A, const __bf16* __restrict__ W,
    const float* __restrict__ bo, const float* __restrict__ x,
    float* __restrict__ out) {
  const int wave = threadIdx.x >> 5;
  const int lane = threadIdx.x & 31;
  const int row0 = (blockIdx.x * 8 + wave) * 16;
  const int col0 = blockIdx.y * 16;
  const __bf16* a_ptr = A + (size_t)row0 * C_DIM;
  const __bf16* w_ptr = W + (size_t)col0 * C_DIM;
  v8f acc = {};
#pragma unroll 4
  for (int kk = 0; kk < C_DIM; kk += 32) {
    __builtin_prefetch(a_ptr + kk + 128, 0, 3);
    v16bf a = load_a16x32(a_ptr + kk, C_DIM);
    v16bf w = load_b_rows(w_ptr + kk, C_DIM);
    acc = __builtin_amdgcn_wmma_f32_16x16x32_bf16(false, a, false, w,
                                                  (short)0, acc, false, false);
  }
  const int hi = lane >> 4, nc = lane & 15;
  const int co = col0 + nc;
  const float bv = bo[co];
#pragma unroll
  for (int r = 0; r < 8; ++r) {
    int gm = row0 + r + hi * 8;        // global row: b*N + n
    int bb = gm >> 12;                 // /4096
    int n  = gm & (N_SP - 1);
    size_t xi = ((size_t)(bb * C_DIM + co)) * N_SP + n;
    out[xi] = x[xi] + acc[r] + bv;
  }
}

// ---------------------------------------------------------------------------
extern "C" void kernel_launch(void* const* d_in, const int* in_sizes, int n_in,
                              void* d_out, int out_size, void* d_ws, size_t ws_size,
                              hipStream_t stream) {
  const float* x     = (const float*)d_in[0];
  const float* gamma = (const float*)d_in[1];
  const float* beta  = (const float*)d_in[2];
  const float* wq    = (const float*)d_in[3];
  const float* bq    = (const float*)d_in[4];
  const float* wk    = (const float*)d_in[5];
  const float* bk    = (const float*)d_in[6];
  const float* wv    = (const float*)d_in[7];
  const float* bv    = (const float*)d_in[8];
  const float* wo    = (const float*)d_in[9];
  const float* bo    = (const float*)d_in[10];
  float* out = (float*)d_out;

  // workspace carve-up
  char* ws = (char*)d_ws;
  float* stats = (float*)ws;                ws += 1024;                       // 128*2 f32
  const size_t act_bytes = (size_t)NTOT * C_DIM * sizeof(__bf16);            // 16 MB
  __bf16* h  = (__bf16*)ws; ws += act_bytes;
  __bf16* qd = (__bf16*)ws; ws += act_bytes;
  __bf16* kd = (__bf16*)ws; ws += act_bytes;
  __bf16* vd = (__bf16*)ws; ws += act_bytes;
  __bf16* ao = (__bf16*)ws; ws += act_bytes;
  const int wn = C_DIM * C_DIM;
  __bf16* wqb = (__bf16*)ws; ws += (size_t)wn * sizeof(__bf16);
  __bf16* wkb = (__bf16*)ws; ws += (size_t)wn * sizeof(__bf16);
  __bf16* wvb = (__bf16*)ws; ws += (size_t)wn * sizeof(__bf16);
  __bf16* wob = (__bf16*)ws; ws += (size_t)wn * sizeof(__bf16);

  // 1) GroupNorm
  gn_stats_kernel<<<BATCH * NGRP, 256, 0, stream>>>(x, stats);
  gn_apply_kernel<<<dim3(N_SP / 32, C_DIM / 32, BATCH), 256, 0, stream>>>(
      x, stats, gamma, beta, h);

  // 2) weights to bf16
  cvt_bf16_kernel<<<(wn + 255) / 256, 256, 0, stream>>>(wq, wqb, wn);
  cvt_bf16_kernel<<<(wn + 255) / 256, 256, 0, stream>>>(wk, wkb, wn);
  cvt_bf16_kernel<<<(wn + 255) / 256, 256, 0, stream>>>(wv, wvb, wn);
  cvt_bf16_kernel<<<(wn + 255) / 256, 256, 0, stream>>>(wo, wob, wn);

  // 3) Q/K/V projections
  dim3 ggrid(NTOT / 128, C_DIM / 16);
  gemm_qkv_kernel<<<ggrid, 256, 0, stream>>>(h, wqb, bq, qd);
  gemm_qkv_kernel<<<ggrid, 256, 0, stream>>>(h, wkb, bk, kd);
  gemm_qkv_kernel<<<ggrid, 256, 0, stream>>>(h, wvb, bv, vd);

  // 4) flash attention (TDM-staged V tiles)
  const float scale = 0.044194173824159216f;  // 1/sqrt(512)
  attn_kernel<<<dim3(N_SP / 16, BATCH), 256, 0, stream>>>(qd, kd, vd, ao, scale);

  // 5) output projection + residual
  oproj_kernel<<<ggrid, 256, 0, stream>>>(ao, wob, bo, x, out);
}
